// Enconder_89524298318133
// MI455X (gfx1250) — matrix operand order repaired and checked by reading, not compiled, not run on hardware
//
#include <hip/hip_runtime.h>
#include <hip/hip_bf16.h>

typedef __attribute__((ext_vector_type(16))) __bf16 v16bf;
typedef __attribute__((ext_vector_type(8)))  float  v8f;
typedef __attribute__((ext_vector_type(4)))  float  f32x4;

#define DIM     768
#define HEADS   12
#define N_TOK   4096
#define BATCH   16
#define QK_ROWS 1536
#define SCALE   0.125f
#define EPS_BN  1e-5f
#define EPS_SQ  1e-5f
#define LSTR    72            // padded LDS stride (bf16 elems), 144B: 16B-aligned, bank-friendly
#define OUT_ELEMS 50331648    // 16*768*4096

// ---- WMMA helpers -----------------------------------------------------------

__device__ __forceinline__ v8f wmma_bf16(v16bf a, v16bf b, v8f c) {
  return __builtin_amdgcn_wmma_f32_16x16x32_bf16(false, a, false, b, (short)0, c, false, false);
}

// A fragment (16x32, M x K): lane holds row m=lane%16; K runs {8h..8h+7, 16+8h..16+8h+7}
__device__ __forceinline__ v16bf frag_a(const __bf16* p0) {
  int lane = threadIdx.x & 31;
  const __bf16* p = p0 + (lane & 15) * LSTR + ((lane >> 4) << 3);
  union { uint4 u[2]; v16bf v; } t;
  t.u[0] = *(const uint4*)(p);
  t.u[1] = *(const uint4*)(p + 16);
  return t.v;
}

// B fragment (32x16, K x N) loaded from row-major [n][k] tile (i.e. B^T row-major):
// lane holds column n=lane%16; K runs {8h..8h+7, 16+8h..16+8h+7}
__device__ __forceinline__ v16bf frag_b(const __bf16* p0) {
  int lane = threadIdx.x & 31;
  const __bf16* p = p0 + (lane & 15) * LSTR + ((lane >> 4) << 3);
  union { uint4 u[2]; v16bf v; } t;
  t.u[0] = *(const uint4*)(p);
  t.u[1] = *(const uint4*)(p + 16);
  return t.v;
}

// ---- BatchNorm path ---------------------------------------------------------

// per-(b,c) sum and sumsq over 4096 tokens
__global__ __launch_bounds__(256) void sumx_kernel(const float* __restrict__ x,
                                                   float* __restrict__ sX,
                                                   float* __restrict__ sQ) {
  __shared__ float s1[256], s2[256];
  int c = blockIdx.x, b = blockIdx.y, t = threadIdx.x;
  const float* p = x + ((size_t)(b * DIM + c) << 12);
  float sum = 0.f, sq = 0.f;
  for (int i = t; i < N_TOK; i += 256) { float v = p[i]; sum += v; sq += v * v; }
  s1[t] = sum; s2[t] = sq; __syncthreads();
  for (int o = 128; o > 0; o >>= 1) {
    if (t < o) { s1[t] += s1[t + o]; s2[t] += s2[t + o]; }
    __syncthreads();
  }
  if (t == 0) { sX[b * DIM + c] = s1[0]; sQ[b * DIM + c] = s2[0]; }
}

__global__ void bn_finalize_kernel(const float* __restrict__ sX, const float* __restrict__ sQ,
                                   float* __restrict__ mean, float* __restrict__ rsig) {
  int c = blockIdx.x * 256 + threadIdx.x;
  if (c >= DIM) return;
  float s = 0.f, q = 0.f;
  for (int b = 0; b < BATCH; ++b) { s += sX[b * DIM + c]; q += sQ[b * DIM + c]; }
  float m = s * (1.f / 65536.f);
  float v = q * (1.f / 65536.f) - m * m;
  mean[c] = m;
  rsig[c] = rsqrtf(v + EPS_BN);
}

__global__ __launch_bounds__(256) void bn_apply_kernel(const float4* __restrict__ x4,
                                                       const float* __restrict__ mean,
                                                       const float* __restrict__ rsig,
                                                       const float* __restrict__ gamma,
                                                       const float* __restrict__ beta,
                                                       float* __restrict__ out) {
  size_t tid = (size_t)blockIdx.x * 256 + threadIdx.x;     // exactly one float4 each
  int c = (int)((tid >> 10) % DIM);
  float m = mean[c], rs = rsig[c], g = gamma[c], bb = beta[c];
  float4 v = x4[tid];
  f32x4 o;
  o.x = fmaxf((v.x - m) * rs * g + bb, 0.f);
  o.y = fmaxf((v.y - m) * rs * g + bb, 0.f);
  o.z = fmaxf((v.z - m) * rs * g + bb, 0.f);
  o.w = fmaxf((v.w - m) * rs * g + bb, 0.f);
  __builtin_nontemporal_store(o, (f32x4*)(out) + tid);     // keep 201MB stream out of L2
}

// ---- Attention path ---------------------------------------------------------

// qkv_w rows 0..1535 (q then k blocks) -> bf16
__global__ void wconv_kernel(const float* __restrict__ w, __bf16* __restrict__ wbf) {
  int i = blockIdx.x * 256 + threadIdx.x;                  // 1536*768 exact
  wbf[i] = (__bf16)w[i];
}

// sb[b][o] = sum_c qkv_w[o][c] * sX[b][c]   (o in [0,1536))
__global__ __launch_bounds__(256) void sbproj_kernel(const float* __restrict__ w,
                                                     const float* __restrict__ sX,
                                                     float* __restrict__ sb) {
  __shared__ float s[256];
  int o = blockIdx.x, b = blockIdx.y, t = threadIdx.x;
  const float* wr = w + (size_t)o * DIM;
  const float* sr = sX + b * DIM;
  float sum = 0.f;
  for (int c = t; c < DIM; c += 256) sum += wr[c] * sr[c];
  s[t] = sum; __syncthreads();
  for (int r = 128; r > 0; r >>= 1) { if (t < r) s[t] += s[t + r]; __syncthreads(); }
  if (t == 0) sb[b * QK_ROWS + o] = s[0];
}

// G[b] = X[b] * X[b]^T  (768x768 per batch, bf16 output), 64x64 tile per workgroup
__global__ __launch_bounds__(256) void gram_kernel(const float* __restrict__ x,
                                                   __bf16* __restrict__ g) {
  __shared__ __align__(16) __bf16 xa[64 * LSTR];
  __shared__ __align__(16) __bf16 xb[64 * LSTR];
  int tile = blockIdx.x, b = blockIdx.y;
  int ti = tile / 12, tj = tile % 12;
  int t = threadIdx.x, wave = t >> 5, lane = t & 31;
  int r = t >> 2, seg = (t & 3) << 4;
  bool diag = (ti == tj);
  const __bf16* xbp = diag ? xa : xb;
  const float* rowA = x + ((size_t)(b * DIM + ti * 64 + r) << 12) + seg;
  const float* rowB = x + ((size_t)(b * DIM + tj * 64 + r) << 12) + seg;

  int st0 = wave * 2, st1 = st0 + 1;
  int mi0 = st0 >> 2, ni0 = st0 & 3, mi1 = st1 >> 2, ni1 = st1 & 3;
  v8f acc0 = {}, acc1 = {};

  for (int n0 = 0; n0 < N_TOK; n0 += 64) {
    __syncthreads();
    {
      const float4* pa = (const float4*)(rowA + n0);
      float4 v0 = pa[0], v1 = pa[1], v2 = pa[2], v3 = pa[3];
      float tmp[16] = {v0.x, v0.y, v0.z, v0.w, v1.x, v1.y, v1.z, v1.w,
                       v2.x, v2.y, v2.z, v2.w, v3.x, v3.y, v3.z, v3.w};
      __bf16* d = xa + r * LSTR + seg;
      #pragma unroll
      for (int i = 0; i < 16; ++i) d[i] = (__bf16)tmp[i];
      if (!diag) {
        const float4* pb = (const float4*)(rowB + n0);
        float4 w0 = pb[0], w1 = pb[1], w2 = pb[2], w3 = pb[3];
        float tm2[16] = {w0.x, w0.y, w0.z, w0.w, w1.x, w1.y, w1.z, w1.w,
                         w2.x, w2.y, w2.z, w2.w, w3.x, w3.y, w3.z, w3.w};
        __bf16* d2 = xb + r * LSTR + seg;
        #pragma unroll
        for (int i = 0; i < 16; ++i) d2[i] = (__bf16)tm2[i];
      }
    }
    __syncthreads();
    #pragma unroll
    for (int kk = 0; kk < 64; kk += 32) {
      acc0 = wmma_bf16(frag_a(xa + mi0 * 16 * LSTR + kk), frag_b(xbp + ni0 * 16 * LSTR + kk), acc0);
      acc1 = wmma_bf16(frag_a(xa + mi1 * 16 * LSTR + kk), frag_b(xbp + ni1 * 16 * LSTR + kk), acc1);
    }
  }
  int n = lane & 15, half = lane >> 4;
  size_t gbase = (size_t)b * DIM * DIM;
  #pragma unroll
  for (int rr = 0; rr < 8; ++rr) {
    int m = rr + half * 8;
    g[gbase + (size_t)(ti * 64 + mi0 * 16 + m) * DIM + (tj * 64 + ni0 * 16 + n)] = (__bf16)acc0[rr];
    g[gbase + (size_t)(ti * 64 + mi1 * 16 + m) * DIM + (tj * 64 + ni1 * 16 + n)] = (__bf16)acc1[rr];
  }
}

// logits = Wq_h * G_b * Wk_h^T, then bias terms, signed-sqrt, softmax -> attn out
__global__ __launch_bounds__(256) void attn_kernel(const __bf16* __restrict__ wbf,
                                                   const __bf16* __restrict__ g,
                                                   const float* __restrict__ qkv_b,
                                                   const float* __restrict__ sb,
                                                   float* __restrict__ attn_out) {
  __shared__ __align__(16) __bf16 wq_t[64 * LSTR];
  __shared__ __align__(16) __bf16 g_t [64 * LSTR];
  __shared__ __align__(16) __bf16 wk_t[64 * LSTR];
  __shared__ __align__(16) __bf16 t_bf[64 * LSTR];
  __shared__ float lg[64 * 65];
  int h = blockIdx.x, b = blockIdx.y;
  int t = threadIdx.x, wave = t >> 5, lane = t & 31;
  int r = t >> 2, seg = (t & 3) << 4;
  int st0 = wave * 2, st1 = st0 + 1;
  int mi0 = st0 >> 2, ni0 = st0 & 3, mi1 = st1 >> 2, ni1 = st1 & 3;
  v8f accL0 = {}, accL1 = {};

  for (int cp = 0; cp < 12; ++cp) {
    __syncthreads();
    { // Wk chunk: wk_t[e][c'] = W[768 + h*64 + e][cp*64 + c']
      const uint4* src = (const uint4*)(wbf + (size_t)(DIM + h * 64 + r) * DIM + cp * 64 + seg);
      uint4* dst = (uint4*)(wk_t + r * LSTR + seg);
      dst[0] = src[0]; dst[1] = src[1];
    }
    v8f accT0 = {}, accT1 = {};
    for (int cc = 0; cc < 12; ++cc) {
      __syncthreads();
      { // Wq chunk: wq_t[d][c] = W[h*64 + d][cc*64 + c]
        const uint4* src = (const uint4*)(wbf + (size_t)(h * 64 + r) * DIM + cc * 64 + seg);
        uint4* dst = (uint4*)(wq_t + r * LSTR + seg);
        dst[0] = src[0]; dst[1] = src[1];
      }
      { // G chunk via symmetry: g_t[c'][c] = G[cp*64+c'][cc*64+c] = B[c][c']
        const uint4* src = (const uint4*)(g + ((size_t)b * DIM + cp * 64 + r) * DIM + cc * 64 + seg);
        uint4* dst = (uint4*)(g_t + r * LSTR + seg);
        dst[0] = src[0]; dst[1] = src[1];
      }
      __syncthreads();
      #pragma unroll
      for (int kk = 0; kk < 64; kk += 32) {
        accT0 = wmma_bf16(frag_a(wq_t + mi0 * 16 * LSTR + kk), frag_b(g_t + ni0 * 16 * LSTR + kk), accT0);
        accT1 = wmma_bf16(frag_a(wq_t + mi1 * 16 * LSTR + kk), frag_b(g_t + ni1 * 16 * LSTR + kk), accT1);
      }
    }
    __syncthreads();
    { // T = Wq*G chunk -> bf16 tile (row-major [d][c'])
      int n = lane & 15, half = lane >> 4;
      #pragma unroll
      for (int rr = 0; rr < 8; ++rr) {
        t_bf[(mi0 * 16 + rr + half * 8) * LSTR + ni0 * 16 + n] = (__bf16)accT0[rr];
        t_bf[(mi1 * 16 + rr + half * 8) * LSTR + ni1 * 16 + n] = (__bf16)accT1[rr];
      }
    }
    __syncthreads();
    #pragma unroll
    for (int kk = 0; kk < 64; kk += 32) {
      accL0 = wmma_bf16(frag_a(t_bf + mi0 * 16 * LSTR + kk), frag_b(wk_t + ni0 * 16 * LSTR + kk), accL0);
      accL1 = wmma_bf16(frag_a(t_bf + mi1 * 16 * LSTR + kk), frag_b(wk_t + ni1 * 16 * LSTR + kk), accL1);
    }
  }
  __syncthreads();
  { // dump raw logits to LDS (fp32)
    int n = lane & 15, half = lane >> 4;
    #pragma unroll
    for (int rr = 0; rr < 8; ++rr) {
      lg[(mi0 * 16 + rr + half * 8) * 65 + ni0 * 16 + n] = accL0[rr];
      lg[(mi1 * 16 + rr + half * 8) * 65 + ni1 * 16 + n] = accL1[rr];
    }
  }
  __syncthreads();
  if (t < 64) { // one row per thread: bias terms, signed sqrt, softmax over e
    int d = t;
    float bq = qkv_b[h * 64 + d];
    float sq = sb[b * QK_ROWS + h * 64 + d];
    float mx = -1e30f;
    for (int e = 0; e < 64; ++e) {
      float bk = qkv_b[DIM + h * 64 + e];
      float sk = sb[b * QK_ROWS + DIM + h * 64 + e];
      float a = SCALE * (lg[d * 65 + e] + bq * sk + sq * bk + 4096.0f * bq * bk);
      float sg = (a > 0.f) ? 1.f : ((a < 0.f) ? -1.f : 0.f);
      float v = sg * sqrtf(fabsf(a) + EPS_SQ);
      lg[d * 65 + e] = v;
      mx = fmaxf(mx, v);
    }
    float sum = 0.f;
    for (int e = 0; e < 64; ++e) sum += __expf(lg[d * 65 + e] - mx);
    float inv = 1.f / sum;
    float* dst = attn_out + (((size_t)(b * HEADS + h) * 64 + d) << 6);
    for (int e = 0; e < 64; ++e) dst[e] = __expf(lg[d * 65 + e] - mx) * inv;
  }
}

// ---- launcher ---------------------------------------------------------------

extern "C" void kernel_launch(void* const* d_in, const int* in_sizes, int n_in,
                              void* d_out, int out_size, void* d_ws, size_t ws_size,
                              hipStream_t stream) {
  (void)in_sizes; (void)n_in; (void)out_size; (void)ws_size;
  const float* x     = (const float*)d_in[0];
  const float* qkv_w = (const float*)d_in[1];
  const float* qkv_b = (const float*)d_in[2];
  const float* gamma = (const float*)d_in[3];
  const float* beta  = (const float*)d_in[4];
  float* out = (float*)d_out;

  char* ws = (char*)d_ws;                         // all offsets 16B-aligned
  float*  mean = (float*)(ws + 0);                //   768 f32
  float*  rsig = (float*)(ws + 3072);             //   768 f32
  float*  sX   = (float*)(ws + 6144);             // 16*768 f32
  float*  sQ   = (float*)(ws + 55296);            // 16*768 f32
  float*  sb   = (float*)(ws + 104448);           // 16*1536 f32
  __bf16* wbf  = (__bf16*)(ws + 202752);          // 1536*768 bf16
  __bf16* gbf  = (__bf16*)(ws + 2562048);         // 16*768*768 bf16 (~19MB)

  sumx_kernel<<<dim3(DIM, BATCH), 256, 0, stream>>>(x, sX, sQ);
  bn_finalize_kernel<<<3, 256, 0, stream>>>(sX, sQ, mean, rsig);
  bn_apply_kernel<<<49152, 256, 0, stream>>>((const float4*)x, mean, rsig, gamma, beta, out);
  wconv_kernel<<<4608, 256, 0, stream>>>(qkv_w, wbf);
  sbproj_kernel<<<dim3(QK_ROWS, BATCH), 256, 0, stream>>>(qkv_w, sX, sb);
  gram_kernel<<<dim3(144, BATCH), 256, 0, stream>>>(x, gbf);
  attn_kernel<<<dim3(HEADS, BATCH), 256, 0, stream>>>(wbf, gbf, qkv_b, sb, out + OUT_ELEMS);
}
